// CausalAttention_2765958938909
// MI455X (gfx1250) — matrix operand order repaired and checked by reading, not compiled, not run on hardware
//
#include <hip/hip_runtime.h>
#include <hip/hip_bf16.h>

// ---------------------------------------------------------------------------
// CausalAttention for MI455X (gfx1250): bf16 WMMA (v_wmma_f32_16x16x32_bf16)
// Shapes: B=8, S=1024, C=512, NH=8, DH=64, CIN=256
// ---------------------------------------------------------------------------

typedef __attribute__((ext_vector_type(16))) __bf16 v16bf;
typedef __attribute__((ext_vector_type(8)))  __bf16 v8bf;
typedef __attribute__((ext_vector_type(8)))  float  v8f;

#define NH   8
#define DH   64
#define SS   1024
#define NB   8
#define CC   512
#define CIN  256

static __device__ __forceinline__ v16bf cat8(v8bf lo, v8bf hi) {
    return __builtin_shufflevector(lo, hi, 0,1,2,3,4,5,6,7,8,9,10,11,12,13,14,15);
}

// ---------------------------------------------------------------------------
// Stage 1: fold weight-norm (W = g * v / ||v||_row) into a bf16 weight matrix
// ---------------------------------------------------------------------------
__global__ __launch_bounds__(32)
void wnorm_kernel(const float* __restrict__ v, const float* __restrict__ g,
                  __bf16* __restrict__ W)
{
    int row = blockIdx.x, lane = threadIdx.x;
    const float* vr = v + (size_t)row * CIN;
    float ss = 0.f;
    for (int i = lane; i < CIN; i += 32) { float x = vr[i]; ss += x * x; }
    #pragma unroll
    for (int off = 16; off >= 1; off >>= 1) ss += __shfl_xor(ss, off, 32);
    float sc = g[row] * rsqrtf(ss);
    for (int i = lane; i < CIN; i += 32)
        W[(size_t)row * CIN + i] = (__bf16)(vr[i] * sc);
}

// ---------------------------------------------------------------------------
// Stage 2: projection GEMM  Y[s][cout] = sum_cin X[s][cin] * W[cout][cin] + b
// Block = 16 tokens; 8 waves x 4 cout tiles. A from LDS-transposed X tile
// (loaded once, reused for all tiles); all 16 B-fragment loads for a tile are
// issued up-front so WMMAs overlap the loads.
// CHANMAJOR=0 -> out[bh][s][dh] (Q,K);  CHANMAJOR=1 -> out[bh][dh][s] (V).
// ---------------------------------------------------------------------------
template <int CHANMAJOR>
__global__ __launch_bounds__(256)
void proj_kernel(const float* __restrict__ x,     // [B][CIN][S]
                 const __bf16* __restrict__ W,    // [CC][CIN] bf16
                 const float* __restrict__ bias,  // [CC]
                 __bf16* __restrict__ out)
{
    __shared__ __align__(16) __bf16 Xs[16][CIN + 8];   // stride 264 (528B rows)

    int bx  = blockIdx.x;
    int b   = bx >> 6;            // S/16 = 64 token tiles per batch
    int s0  = (bx & 63) << 4;
    int tid = threadIdx.x;

    // stage + transpose X tile: thread t loads 16 contiguous floats of cin=t
    {
        int cin = tid;
        const float4* src =
            reinterpret_cast<const float4*>(x + ((size_t)(b * CIN + cin)) * SS + s0);
        #pragma unroll
        for (int j = 0; j < 4; ++j) {
            float4 q = src[j];
            Xs[4*j + 0][cin] = (__bf16)q.x;
            Xs[4*j + 1][cin] = (__bf16)q.y;
            Xs[4*j + 2][cin] = (__bf16)q.z;
            Xs[4*j + 3][cin] = (__bf16)q.w;
        }
    }
    __syncthreads();

    int wave = tid >> 5, lane = tid & 31, hi = lane >> 4, n16 = lane & 15;

    // preload all 8 A fragments (shared across the 4 cout tiles of this wave)
    v16bf afr[8];
    #pragma unroll
    for (int k = 0; k < 8; ++k) {
        int k0 = k * 32;
        v8bf alo = *reinterpret_cast<const v8bf*>(&Xs[n16][k0 + hi * 8]);
        v8bf ahi = *reinterpret_cast<const v8bf*>(&Xs[n16][k0 + 16 + hi * 8]);
        afr[k] = cat8(alo, ahi);
    }

    for (int t = 0; t < 4; ++t) {
        int cout0 = (wave * 4 + t) * 16;

        // issue ALL B-fragment loads for this tile up-front (16 x b128)
        const __bf16* wr = W + (size_t)(cout0 + n16) * CIN + hi * 8;
        v16bf bfr[8];
        #pragma unroll
        for (int k = 0; k < 8; ++k) {
            v8bf blo = *reinterpret_cast<const v8bf*>(wr + 32 * k);
            v8bf bhi = *reinterpret_cast<const v8bf*>(wr + 32 * k + 16);
            bfr[k] = cat8(blo, bhi);
        }

        union { v8f v; float f[8]; } acc;
        #pragma unroll
        for (int r = 0; r < 8; ++r) acc.f[r] = 0.f;
        #pragma unroll
        for (int k = 0; k < 8; ++k)
            acc.v = __builtin_amdgcn_wmma_f32_16x16x32_bf16(
                        false, afr[k], false, bfr[k], (short)0, acc.v, false, false);

        float bb = bias[cout0 + n16];
        int c = cout0 + n16;
        int h = c >> 6, d = c & 63;
        int bh = b * NH + h;
        #pragma unroll
        for (int r = 0; r < 8; ++r) {
            int s = s0 + r + hi * 8;           // D layout: VGPR r -> row r (+8 hi)
            float y = acc.f[r] + bb;
            if (CHANMAJOR)
                out[((size_t)(bh * DH + d)) * SS + s] = (__bf16)y;
            else
                out[((size_t)(bh * SS + s)) * DH + d] = (__bf16)y;
        }
    }
}

// ---------------------------------------------------------------------------
// Stage 3: flash attention, strictly-causal (j < i), one wave per 16-row tile.
// K and V fragments for the whole 32-key step are loaded before the QK^T
// WMMAs + softmax VALU block, so global latency hides under compute.
// ---------------------------------------------------------------------------
__global__ __launch_bounds__(32)
void attn_kernel(const __bf16* __restrict__ Q,   // [bh][s][dh]
                 const __bf16* __restrict__ K,   // [bh][s][dh]
                 const __bf16* __restrict__ V,   // [bh][dh][s]
                 float* __restrict__ out)        // [b][c][s]
{
    __shared__ __align__(16) __bf16 Ps[16][40];  // P staging, padded stride

    int qs = blockIdx.x << 4;
    int bh = blockIdx.y;
    int b  = bh >> 3, h = bh & 7;
    int lane = threadIdx.x, hi = lane >> 4, n16 = lane & 15;

    // Q A-fragments for the two dh-chunks (K-dim = dh, 2 x 32)
    const __bf16* qrow = Q + ((size_t)(bh * SS + qs + n16)) * DH;
    v16bf qa[2];
    #pragma unroll
    for (int cc = 0; cc < 2; ++cc) {
        int d0 = cc * 32 + hi * 8;
        v8bf lo  = *reinterpret_cast<const v8bf*>(qrow + d0);
        v8bf hi8 = *reinterpret_cast<const v8bf*>(qrow + d0 + 16);
        qa[cc] = cat8(lo, hi8);
    }

    union F8 { v8f v; float f[8]; };
    float mrow[8], lrow[8], corr[8];
    F8 o[4];
    #pragma unroll
    for (int r = 0; r < 8; ++r) { mrow[r] = -3.0e38f; lrow[r] = 0.f; }
    #pragma unroll
    for (int t = 0; t < 4; ++t)
        #pragma unroll
        for (int r = 0; r < 8; ++r) o[t].f[r] = 0.f;

    const float scale = 0.125f;   // 1/sqrt(64)
    const int jend = qs + 16;     // strictly causal: keys j <= qs+14 needed

    for (int j0 = 0; j0 < jend; j0 += 32) {
        if (j0 + 32 < jend) {  // pull next step's K/V toward the caches
            __builtin_prefetch(K + ((size_t)(bh * SS + j0 + 32)) * DH, 0, 1);
            __builtin_prefetch(V + ((size_t)(bh * DH)) * SS + j0 + 32, 0, 1);
        }

        // ---- issue ALL K and V fragment loads for this 32-key step ----
        v16bf kf[2][2];                       // [subtile][dh-chunk]
        #pragma unroll
        for (int sub = 0; sub < 2; ++sub) {
            const __bf16* krow =
                K + ((size_t)(bh * SS + j0 + sub * 16 + n16)) * DH;
            #pragma unroll
            for (int cc = 0; cc < 2; ++cc) {
                const __bf16* kp = krow + cc * 32 + hi * 8;
                v8bf lo = *reinterpret_cast<const v8bf*>(kp);
                v8bf h8 = *reinterpret_cast<const v8bf*>(kp + 16);
                kf[sub][cc] = cat8(lo, h8);
            }
        }
        v16bf vf[4];                          // [dh n-tile]
        #pragma unroll
        for (int t = 0; t < 4; ++t) {
            const __bf16* vp =
                V + ((size_t)(bh * DH + t * 16 + n16)) * SS + j0 + hi * 8;
            v8bf lo = *reinterpret_cast<const v8bf*>(vp);
            v8bf h8 = *reinterpret_cast<const v8bf*>(vp + 16);
            vf[t] = cat8(lo, h8);
        }

        // ---- S = Q * K^T for two 16-col subtiles ----
        F8 st[2];
        #pragma unroll
        for (int sub = 0; sub < 2; ++sub) {
            #pragma unroll
            for (int r = 0; r < 8; ++r) st[sub].f[r] = 0.f;
            #pragma unroll
            for (int cc = 0; cc < 2; ++cc)
                st[sub].v = __builtin_amdgcn_wmma_f32_16x16x32_bf16(
                                false, qa[cc], false, kf[sub][cc], (short)0,
                                st[sub].v, false, false);
        }

        // ---- online softmax over this 32-key step ----
        #pragma unroll
        for (int r = 0; r < 8; ++r) {
            int i  = qs + r + hi * 8;            // C-layout: VGPR r -> row r(+8)
            int jA = j0 + n16, jB = j0 + 16 + n16;
            float vA = st[0].f[r] * scale;
            float vB = st[1].f[r] * scale;
            bool  mA = jA < i, mB = jB < i;
            float rm = fmaxf(mA ? vA : -3.0e38f, mB ? vB : -3.0e38f);
            #pragma unroll
            for (int off = 8; off >= 1; off >>= 1)
                rm = fmaxf(rm, __shfl_xor(rm, off, 32));   // 16-lane row reduce
            float nm = fmaxf(mrow[r], rm);
            float pA = mA ? __expf(vA - nm) : 0.f;
            float pB = mB ? __expf(vB - nm) : 0.f;
            float rs = pA + pB;
            #pragma unroll
            for (int off = 8; off >= 1; off >>= 1)
                rs += __shfl_xor(rs, off, 32);
            float cr = __expf(mrow[r] - nm);
            lrow[r] = lrow[r] * cr + rs;
            mrow[r] = nm;
            corr[r] = cr;
            int m = r + hi * 8;                   // stage P in A-friendly layout
            Ps[m][n16]      = (__bf16)pA;
            Ps[m][n16 + 16] = (__bf16)pB;
        }
        __syncthreads();   // single-wave WG: near-free, orders LDS

        // ---- O = O*corr + P * V  (K-dim = 32 keys); V already in registers ----
        v8bf plo = *reinterpret_cast<const v8bf*>(&Ps[n16][hi * 8]);
        v8bf phi = *reinterpret_cast<const v8bf*>(&Ps[n16][16 + hi * 8]);
        v16bf pa = cat8(plo, phi);
        #pragma unroll
        for (int t = 0; t < 4; ++t) {
            #pragma unroll
            for (int r = 0; r < 8; ++r) o[t].f[r] *= corr[r];
            o[t].v = __builtin_amdgcn_wmma_f32_16x16x32_bf16(
                         false, pa, false, vf[t], (short)0, o[t].v, false, false);
        }
        __syncthreads();
    }

    // ---- finalize: divide by row sum; row 0 has l==0 -> zero (start_mask) ----
    #pragma unroll
    for (int r = 0; r < 8; ++r) {
        int s = qs + r + hi * 8;
        float inv = lrow[r] > 0.f ? 1.0f / lrow[r] : 0.f;
        #pragma unroll
        for (int t = 0; t < 4; ++t) {
            int c = h * DH + t * 16 + n16;
            out[((size_t)(b * CC + c)) * SS + s] = o[t].f[r] * inv;
        }
    }
}

// ---------------------------------------------------------------------------
extern "C" void kernel_launch(void* const* d_in, const int* in_sizes, int n_in,
                              void* d_out, int out_size, void* d_ws, size_t ws_size,
                              hipStream_t stream)
{
    const float* query = (const float*)d_in[0];
    const float* key   = (const float*)d_in[1];
    const float* vq = (const float*)d_in[2];
    const float* gq = (const float*)d_in[3];
    const float* bq = (const float*)d_in[4];
    const float* vk = (const float*)d_in[5];
    const float* gk = (const float*)d_in[6];
    const float* bk = (const float*)d_in[7];
    const float* vv = (const float*)d_in[8];
    const float* gv = (const float*)d_in[9];
    const float* bv = (const float*)d_in[10];
    float* out = (float*)d_out;

    // workspace layout (bf16): Wq|Wk|Wv (512x256 each), Q|K|V (8*8*1024*64 each)
    __bf16* Wq = (__bf16*)d_ws;
    __bf16* Wk = Wq + (size_t)CC * CIN;
    __bf16* Wv = Wk + (size_t)CC * CIN;
    __bf16* Qb = Wv + (size_t)CC * CIN;
    __bf16* Kb = Qb + (size_t)NB * NH * SS * DH;
    __bf16* Vb = Kb + (size_t)NB * NH * SS * DH;

    wnorm_kernel<<<CC, 32, 0, stream>>>(vq, gq, Wq);
    wnorm_kernel<<<CC, 32, 0, stream>>>(vk, gk, Wk);
    wnorm_kernel<<<CC, 32, 0, stream>>>(vv, gv, Wv);

    int projBlocks = NB * (SS / 16);   // 512
    proj_kernel<0><<<projBlocks, 256, 0, stream>>>(query, Wq, bq, Qb);
    proj_kernel<0><<<projBlocks, 256, 0, stream>>>(key,   Wk, bk, Kb);
    proj_kernel<1><<<projBlocks, 256, 0, stream>>>(key,   Wv, bv, Vb);

    attn_kernel<<<dim3(SS / 16, NB * NH), 32, 0, stream>>>(Qb, Kb, Vb, out);
}
